// BinaryHead_49400713838618
// MI455X (gfx1250) — matrix operand order repaired and checked by reading, not compiled, not run on hardware
//
#include <hip/hip_runtime.h>
#include <hip/hip_bf16.h>

typedef __attribute__((ext_vector_type(16))) __bf16 v16bf;
typedef __attribute__((ext_vector_type(8)))  float  v8f;

#define STATS_NB 256

// ---------------- BatchNorm statistics (deterministic 2-stage reduction) ----------------
__global__ __launch_bounds__(256)
void stats_partial(const float* __restrict__ x, int N, int C, float* __restrict__ partial) {
  const int c      = threadIdx.x % C;     // C must divide 256 (C = 64 or 128 here)
  const int grp    = threadIdx.x / C;
  const int groups = 256 / C;
  float s = 0.f, s2 = 0.f;
  for (long long i = (long long)blockIdx.x * groups + grp; i < N;
       i += (long long)gridDim.x * groups) {
    float v = x[i * (long long)C + c];
    s += v; s2 += v * v;
  }
  __shared__ float shs[256];
  __shared__ float shq[256];
  shs[threadIdx.x] = s;
  shq[threadIdx.x] = s2;
  __syncthreads();
  if (grp == 0) {
    for (int g2 = 1; g2 < groups; ++g2) { s += shs[g2 * C + c]; s2 += shq[g2 * C + c]; }
    partial[(size_t)blockIdx.x * 2 * C + c]     = s;
    partial[(size_t)blockIdx.x * 2 * C + C + c] = s2;
  }
}

__global__ void stats_final(const float* __restrict__ partial, int NB, int C, int N,
                            const float* __restrict__ g, const float* __restrict__ beta,
                            float* __restrict__ scale, float* __restrict__ shift) {
  int c = threadIdx.x;
  if (c >= C) return;
  float s = 0.f, s2 = 0.f;
  for (int b = 0; b < NB; ++b) {               // fixed order -> deterministic
    s  += partial[(size_t)b * 2 * C + c];
    s2 += partial[(size_t)b * 2 * C + C + c];
  }
  float mean = s / (float)N;
  float var  = s2 / (float)N - mean * mean;    // biased var, matches jnp.var
  float sc   = g[c] * rsqrtf(var + 1e-4f);
  scale[c] = sc;
  shift[c] = beta[c] - mean * sc;
}

// ---------------- fused (BN+ReLU)? + f32->bf16 convert, with column offset ----------------
template<bool BN>
__global__ void cvt_kernel(const float* __restrict__ x, long long N, int C,
                           const float* __restrict__ scale, const float* __restrict__ shift,
                           __hip_bfloat16* __restrict__ out, int ldOut, int colOff) {
  long long idx = (long long)blockIdx.x * blockDim.x + threadIdx.x;
  if (idx >= N * C) return;
  long long i = idx / C;
  int c = (int)(idx % C);
  float v = x[idx];
  if (BN) v = fmaxf(v * scale[c] + shift[c], 0.f);
  out[i * ldOut + colOff + c] = __float2bfloat16(v);
}

// ---------------- repack W[k][Cin][64] (f32) into WMMA bf16 B-fragments ----------------
// Fragment layout (V_WMMA_F32_16X16X32_BF16 B-matrix 32x16, ISA 7.12.2):
//   lane n = lane&15, element j (0..15): K = (j<8 ? j : j+8) + 8*(lane>>4)
// packed[((k*NKK + kk)*4 + wtile)*512 + lane*16 + j]
__global__ void pack_weights(const float* __restrict__ W, int taps, int Cin,
                             __hip_bfloat16* __restrict__ packed) {
  const int nkk = Cin >> 5;
  long long total = (long long)taps * nkk * 4 * 512;
  long long idx = (long long)blockIdx.x * blockDim.x + threadIdx.x;
  if (idx >= total) return;
  int j    = (int)(idx & 15);
  int lane = (int)((idx >> 4) & 31);
  int w    = (int)((idx >> 9) & 3);
  long long t = idx >> 11;                     // k*nkk + kk
  int kk = (int)(t % nkk);
  int k  = (int)(t / nkk);
  int kb = kk * 32 + (j < 8 ? j : j + 8) + 8 * (lane >> 4);
  int n  = w * 16 + (lane & 15);
  packed[idx] = __float2bfloat16(W[((long long)k * Cin + kb) * 64 + n]);
}

// ---------------- gathered GEMM:  out[i,:] (+= ident) += sum_k feats[nbr[i,k],:] @ W[k] ----------------
// 128 threads = 4 waves. Each wave owns 16 DISTINCT output rows and all 4 col tiles
// (block = 64 rows x 64 cols). Each gathered A fragment feeds 4 WMMAs (4 independent
// accumulator chains), so gather traffic is not duplicated across waves and latency hides.
// A-fragment (16x32 bf16, ISA 7.12.2): lane m=lane&15, hi=lane>>4,
//   element j -> K = (j<8 ? j : j+8) + 8*hi  ==> two contiguous 16B chunks per lane.
template<int CIN, bool GATHER, bool IDENT>
__global__ __launch_bounds__(128)
void gather_gemm(const __hip_bfloat16* __restrict__ feats,
                 const int* __restrict__ nbr, int taps,
                 const __hip_bfloat16* __restrict__ packW,
                 const float* __restrict__ ident,
                 float* __restrict__ out) {
  constexpr int NKK = CIN / 32;
  const int wave = threadIdx.x >> 5;
  const int lane = threadIdx.x & 31;
  const int m    = lane & 15;
  const int hi   = lane >> 4;
  const long long row0 = (long long)blockIdx.x * 64 + wave * 16;

  v8f acc0 = {}, acc1 = {}, acc2 = {}, acc3 = {};
  if (IDENT) {
#pragma unroll
    for (int r = 0; r < 8; ++r) {
      long long base = (row0 + r + 8 * hi) * 64 + m;
      acc0[r] = ident[base];
      acc1[r] = ident[base + 16];
      acc2[r] = ident[base + 32];
      acc3[r] = ident[base + 48];
    }
  }

  for (int k = 0; k < taps; ++k) {
    long long srcRow;
    if (GATHER) srcRow = nbr[(row0 + m) * taps + k];
    else        srcRow = row0 + m;
    const __hip_bfloat16* rowp = feats + srcRow * CIN;
#pragma unroll
    for (int kk = 0; kk < NKK; ++kk) {
      v16bf a;
      ((uint4*)&a)[0] = *(const uint4*)(rowp + kk * 32 + 8 * hi);        // K j=0..7
      ((uint4*)&a)[1] = *(const uint4*)(rowp + kk * 32 + 16 + 8 * hi);   // K j=8..15

      const __hip_bfloat16* btile = packW + (long long)(k * NKK + kk) * 4 * 512 + lane * 16;
      v16bf b;
      ((uint4*)&b)[0] = ((const uint4*)(btile + 0 * 512))[0];
      ((uint4*)&b)[1] = ((const uint4*)(btile + 0 * 512))[1];
      acc0 = __builtin_amdgcn_wmma_f32_16x16x32_bf16(false, a, false, b, (short)0, acc0, false, false);
      ((uint4*)&b)[0] = ((const uint4*)(btile + 1 * 512))[0];
      ((uint4*)&b)[1] = ((const uint4*)(btile + 1 * 512))[1];
      acc1 = __builtin_amdgcn_wmma_f32_16x16x32_bf16(false, a, false, b, (short)0, acc1, false, false);
      ((uint4*)&b)[0] = ((const uint4*)(btile + 2 * 512))[0];
      ((uint4*)&b)[1] = ((const uint4*)(btile + 2 * 512))[1];
      acc2 = __builtin_amdgcn_wmma_f32_16x16x32_bf16(false, a, false, b, (short)0, acc2, false, false);
      ((uint4*)&b)[0] = ((const uint4*)(btile + 3 * 512))[0];
      ((uint4*)&b)[1] = ((const uint4*)(btile + 3 * 512))[1];
      acc3 = __builtin_amdgcn_wmma_f32_16x16x32_bf16(false, a, false, b, (short)0, acc3, false, false);
    }
  }
#pragma unroll
  for (int r = 0; r < 8; ++r) {
    long long base = (row0 + r + 8 * hi) * 64 + m;
    out[base]      = acc0[r];
    out[base + 16] = acc1[r];
    out[base + 32] = acc2[r];
    out[base + 48] = acc3[r];
  }
}

// ---------------- classifier: BN+ReLU+Linear(64,1) ----------------
__global__ void classifier_kernel(const float* __restrict__ x, long long N,
                                  const float* __restrict__ scale, const float* __restrict__ shift,
                                  const float* __restrict__ w, const float* __restrict__ bias,
                                  float* __restrict__ scores) {
  long long i = (long long)blockIdx.x * blockDim.x + threadIdx.x;
  if (i >= N) return;
  float s = bias[0];
#pragma unroll
  for (int c = 0; c < 64; ++c) {
    float h = fmaxf(x[i * 64 + c] * scale[c] + shift[c], 0.f);
    s += h * w[c];
  }
  scores[i] = s;
}

// ---------------- voxel -> point gather + threshold (sigmoid(s)>0.5 <=> s>0) ----------------
__global__ void gather_out(const float* __restrict__ scores, const int* __restrict__ v2p,
                           long long M, float* __restrict__ outScores, int* __restrict__ outPreds) {
  long long mI = (long long)blockIdx.x * blockDim.x + threadIdx.x;
  if (mI >= M) return;
  float s = scores[v2p[mI]];
  outScores[mI] = s;
  outPreds[mI]  = (s > 0.f) ? 1 : 0;
}

extern "C" void kernel_launch(void* const* d_in, const int* in_sizes, int n_in,
                              void* d_out, int out_size, void* d_ws, size_t ws_size,
                              hipStream_t stream) {
  (void)n_in; (void)out_size; (void)ws_size;
  // inputs in setup_inputs() dict order (params flattened depth-first, insertion order)
  const float* x0    = (const float*)d_in[0];
  const float* x1    = (const float*)d_in[1];
  const int*   nbr0  = (const int*)d_in[2];
  const int*   upn   = (const int*)d_in[3];
  const int*   v2p   = (const int*)d_in[4];
  const float* dbn_g = (const float*)d_in[5];
  const float* dbn_b = (const float*)d_in[6];
  const float* dW    = (const float*)d_in[7];
  const float* b0bn1g= (const float*)d_in[8];
  const float* b0bn1b= (const float*)d_in[9];
  const float* b0w1  = (const float*)d_in[10];
  const float* b0bn2g= (const float*)d_in[11];
  const float* b0bn2b= (const float*)d_in[12];
  const float* b0w2  = (const float*)d_in[13];
  const float* b0wi  = (const float*)d_in[14];
  const float* b1bn1g= (const float*)d_in[15];
  const float* b1bn1b= (const float*)d_in[16];
  const float* b1w1  = (const float*)d_in[17];
  const float* b1bn2g= (const float*)d_in[18];
  const float* b1bn2b= (const float*)d_in[19];
  const float* b1w2  = (const float*)d_in[20];
  const float* clsg  = (const float*)d_in[21];
  const float* clsb  = (const float*)d_in[22];
  const float* clsw  = (const float*)d_in[23];
  const float* clsbias = (const float*)d_in[24];

  const long long N0 = in_sizes[0] / 64;    // 200000 (multiple of 64)
  const long long N1 = in_sizes[1] / 128;   // 60000
  const long long M  = in_sizes[4];         // 500000

  // ---------- workspace layout ----------
  char* ws = (char*)d_ws;
  size_t off = 0;
  auto alloc = [&](size_t bytes) -> void* {
    void* p = ws + off;
    off = (off + bytes + 255) & ~(size_t)255;
    return p;
  };
  auto packSz = [](long long taps, int Cin) { return (size_t)taps * (Cin / 32) * 4 * 512; };

  __hip_bfloat16* pDeconv = (__hip_bfloat16*)alloc(packSz(8, 128) * 2);
  __hip_bfloat16* pB0W1   = (__hip_bfloat16*)alloc(packSz(27, 128) * 2);
  __hip_bfloat16* pB0W2   = (__hip_bfloat16*)alloc(packSz(27, 64) * 2);
  __hip_bfloat16* pWi     = (__hip_bfloat16*)alloc(packSz(1, 128) * 2);
  __hip_bfloat16* pB1W1   = (__hip_bfloat16*)alloc(packSz(27, 64) * 2);
  __hip_bfloat16* pB1W2   = (__hip_bfloat16*)alloc(packSz(27, 64) * 2);
  float* partial = (float*)alloc((size_t)STATS_NB * 2 * 128 * 4);
  float* scl0 = (float*)alloc(128 * 4);  float* sft0 = (float*)alloc(128 * 4);
  float* scl1 = (float*)alloc(128 * 4);  float* sft1 = (float*)alloc(128 * 4);
  float* scl2 = (float*)alloc(128 * 4);  float* sft2 = (float*)alloc(128 * 4);
  float* scl3 = (float*)alloc(128 * 4);  float* sft3 = (float*)alloc(128 * 4);
  float* scl4 = (float*)alloc(128 * 4);  float* sft4 = (float*)alloc(128 * 4);
  float* scl5 = (float*)alloc(128 * 4);  float* sft5 = (float*)alloc(128 * 4);
  __hip_bfloat16* h1   = (__hip_bfloat16*)alloc((size_t)N1 * 128 * 2);
  __hip_bfloat16* catb = (__hip_bfloat16*)alloc((size_t)N0 * 128 * 2);
  __hip_bfloat16* hbn  = (__hip_bfloat16*)alloc((size_t)N0 * 128 * 2);
  __hip_bfloat16* h64a = hbn;                 // reuse: hbn dead after block0 conv1
  __hip_bfloat16* h64b = hbn + (size_t)N0 * 64;
  float* yb     = (float*)alloc((size_t)N0 * 64 * 4);
  float* identb = (float*)alloc((size_t)N0 * 64 * 4);
  float* preb   = (float*)alloc((size_t)N0 * 64 * 4);
  float* out0   = (float*)alloc((size_t)N0 * 64 * 4);
  float* out1   = (float*)alloc((size_t)N0 * 64 * 4);
  float* scores = (float*)alloc((size_t)N0 * 4);

  auto gridOf = [](long long n) { return (int)((n + 255) / 256); };
  auto bnstats = [&](const float* x, long long N, int C, const float* g, const float* beta,
                     float* scl, float* sft) {
    stats_partial<<<STATS_NB, 256, 0, stream>>>(x, (int)N, C, partial);
    stats_final<<<1, 128, 0, stream>>>(partial, STATS_NB, C, (int)N, g, beta, scl, sft);
  };

  // ---------- 0) repack all weights to bf16 WMMA fragments ----------
  pack_weights<<<gridOf(packSz(8, 128)), 256, 0, stream>>>(dW, 8, 128, pDeconv);
  pack_weights<<<gridOf(packSz(27, 128)), 256, 0, stream>>>(b0w1, 27, 128, pB0W1);
  pack_weights<<<gridOf(packSz(27, 64)), 256, 0, stream>>>(b0w2, 27, 64, pB0W2);
  pack_weights<<<gridOf(packSz(1, 128)), 256, 0, stream>>>(b0wi, 1, 128, pWi);
  pack_weights<<<gridOf(packSz(27, 64)), 256, 0, stream>>>(b1w1, 27, 64, pB1W1);
  pack_weights<<<gridOf(packSz(27, 64)), 256, 0, stream>>>(b1w2, 27, 64, pB1W2);

  const int gemmGrid = (int)(N0 / 64);   // each block covers 64 output rows

  // ---------- 1) deconv branch: BN+ReLU(x1) -> bf16, gathered GEMM (8 taps, 128->64) ----------
  bnstats(x1, N1, 128, dbn_g, dbn_b, scl0, sft0);
  cvt_kernel<true><<<gridOf(N1 * 128), 256, 0, stream>>>(x1, N1, 128, scl0, sft0, h1, 128, 0);
  gather_gemm<128, true, false><<<gemmGrid, 128, 0, stream>>>(h1, upn, 8, pDeconv, nullptr, yb);

  // ---------- 2) concat [x0 | y] as raw bf16 (for ident) + per-channel BN stats ----------
  cvt_kernel<false><<<gridOf(N0 * 64), 256, 0, stream>>>(x0, N0, 64, nullptr, nullptr, catb, 128, 0);
  cvt_kernel<false><<<gridOf(N0 * 64), 256, 0, stream>>>(yb, N0, 64, nullptr, nullptr, catb, 128, 64);
  bnstats(x0, N0, 64, b0bn1g,      b0bn1b,      scl1,      sft1);       // channels 0..63
  bnstats(yb, N0, 64, b0bn1g + 64, b0bn1b + 64, scl1 + 64, sft1 + 64);  // channels 64..127
  cvt_kernel<true><<<gridOf(N0 * 64), 256, 0, stream>>>(x0, N0, 64, scl1,      sft1,      hbn, 128, 0);
  cvt_kernel<true><<<gridOf(N0 * 64), 256, 0, stream>>>(yb, N0, 64, scl1 + 64, sft1 + 64, hbn, 128, 64);

  // ---------- 3) block0: ident = cat @ wi ; conv1 (27,128->64) ; BN+ReLU ; conv2 (+ident) ----------
  gather_gemm<128, false, false><<<gemmGrid, 128, 0, stream>>>(catb, nullptr, 1, pWi, nullptr, identb);
  gather_gemm<128, true,  false><<<gemmGrid, 128, 0, stream>>>(hbn, nbr0, 27, pB0W1, nullptr, preb);
  bnstats(preb, N0, 64, b0bn2g, b0bn2b, scl2, sft2);
  cvt_kernel<true><<<gridOf(N0 * 64), 256, 0, stream>>>(preb, N0, 64, scl2, sft2, h64a, 64, 0);
  gather_gemm<64, true, true><<<gemmGrid, 128, 0, stream>>>(h64a, nbr0, 27, pB0W2, identb, out0);

  // ---------- 4) block1: BN+ReLU ; conv1 ; BN+ReLU ; conv2 (+identity out0) ----------
  bnstats(out0, N0, 64, b1bn1g, b1bn1b, scl3, sft3);
  cvt_kernel<true><<<gridOf(N0 * 64), 256, 0, stream>>>(out0, N0, 64, scl3, sft3, h64b, 64, 0);
  gather_gemm<64, true, false><<<gemmGrid, 128, 0, stream>>>(h64b, nbr0, 27, pB1W1, nullptr, preb);
  bnstats(preb, N0, 64, b1bn2g, b1bn2b, scl4, sft4);
  cvt_kernel<true><<<gridOf(N0 * 64), 256, 0, stream>>>(preb, N0, 64, scl4, sft4, h64a, 64, 0);
  gather_gemm<64, true, true><<<gemmGrid, 128, 0, stream>>>(h64a, nbr0, 27, pB1W2, out0, out1);

  // ---------- 5) classifier + point gather / threshold ----------
  bnstats(out1, N0, 64, clsg, clsb, scl5, sft5);
  classifier_kernel<<<gridOf(N0), 256, 0, stream>>>(out1, N0, scl5, sft5, clsw, clsbias, scores);
  gather_out<<<gridOf(M), 256, 0, stream>>>(scores, v2p, M,
                                            (float*)d_out, ((int*)d_out) + M);
}
